// QNetFRMQN_32031866094301
// MI455X (gfx1250) — matrix operand order repaired and checked by reading, not compiled, not run on hardware
//
#include <hip/hip_runtime.h>

// ---------------------------------------------------------------------------
// FRMQN Q-network for MI455X (gfx1250, wave32, WMMA, TDM).
//   conv1/2/3: implicit-GEMM WMMA bf16; input image staged in LDS
//              (TENSOR_LOAD_TO_LDS when available, else VGPR staging)
//   GEMM1 (WMMA bf16, ping-pong pipelined): feat @ [w_key; w_val; W_ih_x]^T
//   scan (1 WGP, 32 waves, WMMA per step): GRU + causal attention
//   GEMM2 (WMMA bf16): hseq @ w_hidden^T + oseq, ReLU
//   heads: dueling Q
// ---------------------------------------------------------------------------

typedef __bf16 bf16;
typedef __attribute__((ext_vector_type(16))) __bf16 v16bf;
typedef __attribute__((ext_vector_type(8)))  __bf16 v8bf;
typedef __attribute__((ext_vector_type(8)))  float  v8f;
typedef __attribute__((ext_vector_type(4)))  unsigned int v4u;
typedef __attribute__((ext_vector_type(8)))  int v8i;
typedef __attribute__((ext_vector_type(4)))  int v4i;

#define BB   32
#define SS   64
#define HID  256
#define FEAT 4096
#define NACT 6
#define ROWS (BB * SS)          // 2048
#define G1N  1280               // 256 keys + 256 vals + 768 gi_x

#if __has_builtin(__builtin_amdgcn_tensor_load_to_lds) && \
    __has_builtin(__builtin_amdgcn_s_wait_tensorcnt)
#define HAVE_TDM 1
#endif

// ---------------- WMMA fragment loaders (ISA 7.12.2 layouts) ----------------
// A (16x32 bf16): lane l holds row M=l&15; K chunks (l>>4)*8..+7 and +16..+23
__device__ __forceinline__ v16bf load_a_frag(const bf16* __restrict__ A, int lda,
                                             int m0, int k0, int lane) {
  const bf16* p = A + (size_t)(m0 + (lane & 15)) * lda + k0 + ((lane >> 4) << 3);
  v8bf lo = *(const v8bf*)(p);
  v8bf hi = *(const v8bf*)(p + 16);
  v16bf a;
#pragma unroll
  for (int i = 0; i < 8; ++i) { a[i] = lo[i]; a[8 + i] = hi[i]; }
  return a;
}
// B (32x16 bf16) for C = A @ W^T: lane l holds N=l&15; K chunks (l>>4)*8..+7 and +16..+23 from W row n.
__device__ __forceinline__ v16bf load_b_frag(const bf16* __restrict__ W, int ldb,
                                             int n0, int k0, int lane) {
  const bf16* p = W + (size_t)(n0 + (lane & 15)) * ldb + k0 + ((lane >> 4) << 3);
  v8bf lo = *(const v8bf*)(p);
  v8bf hi = *(const v8bf*)(p + 16);
  v16bf b;
#pragma unroll
  for (int i = 0; i < 8; ++i) { b[i] = lo[i]; b[8 + i] = hi[i]; }
  return b;
}

// ---------------- TDM: contiguous bf16 block, memory -> LDS offset 0 --------
#ifdef HAVE_TDM
__device__ __forceinline__ void tdm_load_1d(const void* src, unsigned nelem2b) {
  unsigned long long va = (unsigned long long)src;
  // D# group 0: count=1 | lds_addr=0 | global_addr | type=2 ("image")
  v4u g0 = {1u, 0u, (unsigned)(va & 0xFFFFFFFFu),
            ((unsigned)((va >> 32) & 0x1FFFFFFu)) | (2u << 30)};
  // D# group 1: data_size=1 (2B); tensor_dim0 = tile_dim0 = nelem2b;
  // tensor_dim1 = tile_dim1 = 1; tensor_dim0_stride = nelem2b.
  v8i g1 = {(int)(1u << 16),
            (int)((nelem2b & 0xFFFFu) << 16),
            (int)(((nelem2b >> 16) & 0xFFFFu) | (1u << 16)),
            (int)((nelem2b & 0xFFFFu) << 16),
            1,
            (int)nelem2b,
            0, 0};
  v4i z4 = {0, 0, 0, 0};
  v8i z8 = {0, 0, 0, 0, 0, 0, 0, 0};
  // 6-arg form (clang-23 / therock-10.0 headers)
  __builtin_amdgcn_tensor_load_to_lds(g0, g1, z4, z4, z8, 0);
}
#endif

// ---------------- generic bf16 WMMA GEMM: C = A @ W^T (+addend, relu) -------
// 256 thr (8 waves); block tile 128(M) x 64(N); wave 16x64.
// K must be a multiple of 64 (ping-pong double buffer, no rotate moves).
__global__ void __launch_bounds__(256)
gemm_bf16_nt(const bf16* __restrict__ A, const bf16* __restrict__ W,
             float* __restrict__ C, const float* __restrict__ addend,
             int M, int N, int K, int relu) {
  const int lane = threadIdx.x & 31;
  const int wave = threadIdx.x >> 5;
  const int m0 = blockIdx.x * 128 + wave * 16;
  const int n0 = blockIdx.y * 64;

  v8f acc[4];
#pragma unroll
  for (int i = 0; i < 4; ++i)
    acc[i] = (v8f){0.f, 0.f, 0.f, 0.f, 0.f, 0.f, 0.f, 0.f};

  v16bf a0 = load_a_frag(A, K, m0, 0, lane);
  v16bf b0[4], b1[4];
#pragma unroll
  for (int i = 0; i < 4; ++i) b0[i] = load_b_frag(W, K, n0 + i * 16, 0, lane);

  for (int k0 = 0; k0 < K; k0 += 64) {
    if (k0 + 1024 < K)
      __builtin_prefetch(A + (size_t)(m0 + (lane & 15)) * K + k0 + 1024, 0, 3);
    // set 1 <- k0+32 while consuming set 0
    v16bf a1 = load_a_frag(A, K, m0, k0 + 32, lane);
#pragma unroll
    for (int i = 0; i < 4; ++i) b1[i] = load_b_frag(W, K, n0 + i * 16, k0 + 32, lane);
#pragma unroll
    for (int i = 0; i < 4; ++i)
      acc[i] = __builtin_amdgcn_wmma_f32_16x16x32_bf16(
          false, a0, false, b0[i], (short)0, acc[i], false, false);
    // set 0 <- k0+64 while consuming set 1 (last iter reloads k=0, discarded)
    const int kn = (k0 + 64 < K) ? (k0 + 64) : 0;
    a0 = load_a_frag(A, K, m0, kn, lane);
#pragma unroll
    for (int i = 0; i < 4; ++i) b0[i] = load_b_frag(W, K, n0 + i * 16, kn, lane);
#pragma unroll
    for (int i = 0; i < 4; ++i)
      acc[i] = __builtin_amdgcn_wmma_f32_16x16x32_bf16(
          false, a1, false, b1[i], (short)0, acc[i], false, false);
  }

  const int rbase = m0 + ((lane >> 4) << 3);
  const int cbase = lane & 15;
#pragma unroll
  for (int i = 0; i < 4; ++i) {
    const int n = n0 + i * 16 + cbase;
#pragma unroll
    for (int r = 0; r < 8; ++r) {
      size_t idx = (size_t)(rbase + r) * N + n;
      float v = acc[i][r];
      if (addend) v += addend[idx];
      if (relu) v = v > 0.f ? v : 0.f;
      C[idx] = v;
    }
  }
}

// ---------------- f32 -> bf16 strided convert (weight packing) --------------
__global__ void cvt_bf16(bf16* __restrict__ dst, const float* __restrict__ src,
                         int rows, int cols, int src_ld, int src_off) {
  int idx = blockIdx.x * blockDim.x + threadIdx.x;
  if (idx >= rows * cols) return;
  int r = idx / cols, c = idx - r * cols;
  dst[(size_t)r * cols + c] = (bf16)src[(size_t)r * src_ld + src_off + c];
}

// ---------------- conv weight im2col pack with zero padding -----------------
template <int IC, int KS, int GRP, int PADP>
__global__ void pack_conv_w(bf16* __restrict__ dst, const float* __restrict__ w,
                            int OC) {
  constexpr int PSZ = KS * KS;
  constexpr int NGRP = (IC + GRP - 1) / GRP;
  constexpr int KTOT = NGRP * PADP;
  int idx = blockIdx.x * blockDim.x + threadIdx.x;
  if (idx >= OC * KTOT) return;
  int oc = idx / KTOT, k = idx - oc * KTOT;
  int g = k / PADP, kk = k - g * PADP;
  int dic = kk / PSZ, r = kk - dic * PSZ;
  int ic = g * GRP + dic;
  float v = 0.f;
  if (dic < GRP && ic < IC) v = w[((size_t)oc * IC + ic) * PSZ + r];
  dst[idx] = (bf16)v;
}

// ---------------- implicit-GEMM WMMA convolution (stride 2, ReLU) -----------
template <bool IN_F32, int IC, int IH, int IW, int KS, int GRP, int PADP,
          int OC, int OH, int OW>
__global__ void __launch_bounds__(256)
conv_wmma(const void* __restrict__ in, const bf16* __restrict__ wp,
          const float* __restrict__ bias, bf16* __restrict__ out) {
  constexpr int PSZ  = KS * KS;
  constexpr int NGRP = (IC + GRP - 1) / GRP;
  constexpr int KTOT = NGRP * PADP;
  constexpr int NPIX = OH * OW;
  constexpr int MT   = (NPIX + 15) / 16;
  constexpr int NT   = OC / 16;
  constexpr int INSZ = IC * IH * IW;

  extern __shared__ __align__(16) bf16 sh_in[];   // INSZ bf16 at LDS offset 0

  const int img = blockIdx.x;
  const int tid = threadIdx.x, lane = tid & 31, wave = tid >> 5;

  // ---- stage input image into LDS ----
  if constexpr (IN_F32) {
    const float* src = (const float*)in + (size_t)img * INSZ;
    for (int i = tid; i < INSZ; i += 256) sh_in[i] = (bf16)src[i];
  } else {
#ifdef HAVE_TDM
    if (wave == 0) {   // one TDM DMA for the whole image; EXEC-independent
      tdm_load_1d((const bf16*)in + (size_t)img * INSZ, (unsigned)INSZ);
      __builtin_amdgcn_s_wait_tensorcnt(0);
    }
#else
    const bf16* src = (const bf16*)in + (size_t)img * INSZ;
    for (int i = tid; i < INSZ; i += 256) sh_in[i] = src[i];
#endif
  }
  __syncthreads();

  for (int mt = wave; mt < MT; mt += 8) {          // wave-uniform branch
    v8f acc[NT];
#pragma unroll
    for (int i = 0; i < NT; ++i)
      acc[i] = (v8f){0.f, 0.f, 0.f, 0.f, 0.f, 0.f, 0.f, 0.f};

    const int p0 = mt * 16 + (lane & 15);
    const int p  = p0 < NPIX ? p0 : NPIX - 1;      // clamp (dup rows discarded)
    const int oy = p / OW, ox = p - (p / OW) * OW;
    const bf16* base = sh_in + (oy * 2) * IW + (ox * 2);

    for (int g = 0; g < NGRP; ++g) {
      const int icbase = g * GRP;
#pragma unroll
      for (int c = 0; c < PADP / 32; ++c) {
        // gather A fragment (16 bf16/lane) from LDS, im2col on the fly
        v16bf a;
        const int kb = c * 32 + ((lane >> 4) << 3);
#pragma unroll
        for (int e = 0; e < 8; ++e) {
#pragma unroll
          for (int h = 0; h < 2; ++h) {
            const int k   = kb + h * 16 + e;
            const int dic = k / PSZ;
            const int kk  = k - dic * PSZ;
            const int ky  = kk / KS, kx = kk - (kk / KS) * KS;
            const bool ok = (dic < GRP) && (icbase + dic < IC);
            const int ic  = ok ? (icbase + dic) : 0;
            bf16 v = base[ic * (IH * IW) + ky * IW + kx];
            a[h * 8 + e] = ok ? v : (bf16)0.f;
          }
        }
#pragma unroll
        for (int i = 0; i < NT; ++i) {
          v16bf b = load_b_frag(wp, KTOT, i * 16, g * PADP + c * 32, lane);
          acc[i] = __builtin_amdgcn_wmma_f32_16x16x32_bf16(
              false, a, false, b, (short)0, acc[i], false, false);
        }
      }
    }
    const int rb = mt * 16 + ((lane >> 4) << 3);
    const int cb = lane & 15;
#pragma unroll
    for (int i = 0; i < NT; ++i) {
      const int oc = i * 16 + cb;
      const float bv = bias[oc];
#pragma unroll
      for (int r = 0; r < 8; ++r) {
        const int pix = rb + r;
        if (pix < NPIX) {
          float v = acc[i][r] + bv;
          v = v > 0.f ? v : 0.f;
          out[(size_t)img * OC * NPIX + (size_t)oc * NPIX + pix] = (bf16)v;
        }
      }
    }
  }
}

// ---------------- 16x16 tile GEMM over K=256, ping-pong (for scan) ----------
__device__ __forceinline__ v8f tile_gemm_256(const bf16* __restrict__ Ash,
                                             const bf16* __restrict__ W,
                                             int m0, int n0, int lane) {
  v8f acc = (v8f){0.f, 0.f, 0.f, 0.f, 0.f, 0.f, 0.f, 0.f};
  v16bf a0 = load_a_frag(Ash, HID, m0, 0, lane);
  v16bf b0 = load_b_frag(W, HID, n0, 0, lane);
#pragma unroll
  for (int k0 = 0; k0 < HID; k0 += 64) {
    v16bf a1 = load_a_frag(Ash, HID, m0, k0 + 32, lane);
    v16bf b1 = load_b_frag(W, HID, n0, k0 + 32, lane);
    acc = __builtin_amdgcn_wmma_f32_16x16x32_bf16(false, a0, false, b0,
                                                  (short)0, acc, false, false);
    const int kn = (k0 + 64 < HID) ? (k0 + 64) : 0;
    a0 = load_a_frag(Ash, HID, m0, kn, lane);
    b0 = load_b_frag(W, HID, n0, kn, lane);
    acc = __builtin_amdgcn_wmma_f32_16x16x32_bf16(false, a1, false, b1,
                                                  (short)0, acc, false, false);
  }
  return acc;
}

// ---------------- sequential GRU + causal-attention scan --------------------
__device__ __forceinline__ float sigmoidf_(float x) { return 1.f / (1.f + __expf(-x)); }

__global__ void __launch_bounds__(1024)
scan_gru_attn(const float* __restrict__ kv,        // [2048,1280] keys|vals|gi_x
              const bf16* __restrict__ wihr,       // [768,256] = W_ih[:,4096:]
              const bf16* __restrict__ whh,        // [768,256]
              const float* __restrict__ b_ih, const float* __restrict__ b_hh,
              bf16* __restrict__ hseq_bf,          // [2048,256]
              float* __restrict__ oseq) {          // [2048,256]
  extern __shared__ __align__(16) char smem[];
  float* sh_h   = (float*)smem;                    // 32*256 f32
  bf16*  sh_hbf = (bf16*)(sh_h + BB * HID);        // 32*256 bf16
  bf16*  sh_rbf = sh_hbf + BB * HID;               // 32*256 bf16
  float* sh_g   = (float*)(sh_rbf + BB * HID);     // 32*768 f32
  float* sh_hn  = sh_g + BB * 768;                 // 32*256 f32
  float* sh_sc  = sh_hn + BB * HID;                // 32*64  f32

  const int tid = threadIdx.x, lane = tid & 31, wave = tid >> 5;

  for (int i = tid; i < BB * HID; i += 1024) {
    sh_h[i] = 0.f; sh_hbf[i] = (bf16)0.f; sh_rbf[i] = (bf16)0.f;
  }
  __syncthreads();

  for (int t = 0; t < SS; ++t) {
    // phase 1: gi = read @ W_ih_r^T + gi_x + b_ih  (96 tiles, 3/wave)
    for (int tile = wave * 3; tile < wave * 3 + 3; ++tile) {
      const int m0 = (tile & 1) * 16, n0 = (tile >> 1) * 16;
      v8f acc = tile_gemm_256(sh_rbf, wihr, m0, n0, lane);
      const int rb = m0 + ((lane >> 4) << 3), n = n0 + (lane & 15);
#pragma unroll
      for (int r = 0; r < 8; ++r) {
        const int m = rb + r;
        sh_g[m * 768 + n] = acc[r] + kv[(size_t)(m * SS + t) * G1N + 512 + n] + b_ih[n];
      }
    }
    __syncthreads();
    // phase 2: gh = hidden @ W_hh^T + b_hh; fold r,z into gi
    for (int tile = wave * 3; tile < wave * 3 + 3; ++tile) {
      const int m0 = (tile & 1) * 16, n0 = (tile >> 1) * 16;
      v8f acc = tile_gemm_256(sh_hbf, whh, m0, n0, lane);
      const int rb = m0 + ((lane >> 4) << 3), n = n0 + (lane & 15);
#pragma unroll
      for (int r = 0; r < 8; ++r) {
        const int m = rb + r;
        float v = acc[r] + b_hh[n];
        if (n < 512) sh_g[m * 768 + n] += v;
        else         sh_hn[m * HID + (n - 512)] = v;
      }
    }
    __syncthreads();
    // phase 3: GRU gates
    for (int i = tid; i < BB * HID; i += 1024) {
      const int m = i >> 8, k = i & 255;
      float r  = sigmoidf_(sh_g[m * 768 + k]);
      float z  = sigmoidf_(sh_g[m * 768 + 256 + k]);
      float nn = tanhf(sh_g[m * 768 + 512 + k] + r * sh_hn[i]);
      float h  = (1.f - z) * nn + z * sh_h[i];
      sh_h[i] = h; sh_hbf[i] = (bf16)h;
      hseq_bf[(size_t)(m * SS + t) * HID + k] = (bf16)h;
    }
    __syncthreads();
    // phase 4: causal scores s<=t
    const int np = BB * (t + 1);
    for (int pp = tid; pp < np; pp += 1024) {
      const int m = pp / (t + 1), s = pp - m * (t + 1);
      const float* kp = kv + (size_t)(m * SS + s) * G1N;
      const float* hp = sh_h + m * HID;
      float d = 0.f;
      for (int k = 0; k < HID; ++k) d += kp[k] * hp[k];
      sh_sc[m * SS + s] = d;
    }
    __syncthreads();
    // phase 5: per-batch softmax + memory read (wave = batch)
    {
      const int m = wave;
      float mx = -3.4e38f;
      for (int s = lane; s <= t; s += 32) mx = fmaxf(mx, sh_sc[m * SS + s]);
#pragma unroll
      for (int o = 16; o > 0; o >>= 1) mx = fmaxf(mx, __shfl_xor(mx, o, 32));
      float sm = 0.f;
      for (int s = lane; s <= t; s += 32) {
        float e = __expf(sh_sc[m * SS + s] - mx);
        sh_sc[m * SS + s] = e; sm += e;
      }
#pragma unroll
      for (int o = 16; o > 0; o >>= 1) sm += __shfl_xor(sm, o, 32);
      const float inv = 1.f / sm;
#pragma unroll
      for (int j = 0; j < 8; ++j) {
        const int k = lane + 32 * j;
        float acc = 0.f;
        for (int s = 0; s <= t; ++s)
          acc += sh_sc[m * SS + s] * kv[(size_t)(m * SS + s) * G1N + 256 + k];
        acc *= inv;
        sh_rbf[m * HID + k] = (bf16)acc;
        oseq[(size_t)(m * SS + t) * HID + k] = acc;
      }
    }
    __syncthreads();
  }
}

// ---------------- dueling heads ---------------------------------------------
__global__ void heads_q(const float* __restrict__ g, const float* __restrict__ adv_w,
                        const float* __restrict__ adv_b, const float* __restrict__ val_w,
                        const float* __restrict__ val_b, float* __restrict__ q) {
  int row = blockIdx.x * blockDim.x + threadIdx.x;
  if (row >= ROWS) return;
  const float* gp = g + (size_t)row * HID;
  float adv[NACT], mean = 0.f;
#pragma unroll
  for (int a = 0; a < NACT; ++a) {
    float s = adv_b[a];
    for (int k = 0; k < HID; ++k) s += gp[k] * adv_w[a * HID + k];
    adv[a] = s; mean += s;
  }
  mean *= (1.f / NACT);
  float v = val_b[0];
  for (int k = 0; k < HID; ++k) v += gp[k] * val_w[k];
#pragma unroll
  for (int a = 0; a < NACT; ++a) q[(size_t)row * NACT + a] = v + adv[a] - mean;
}

// ---------------------------------------------------------------------------
extern "C" void kernel_launch(void* const* d_in, const int* in_sizes, int n_in,
                              void* d_out, int out_size, void* d_ws, size_t ws_size,
                              hipStream_t stream) {
  const float* obs   = (const float*)d_in[0];
  const float* c1w   = (const float*)d_in[1];
  const float* c1b   = (const float*)d_in[2];
  const float* c2w   = (const float*)d_in[3];
  const float* c2b   = (const float*)d_in[4];
  const float* c3w   = (const float*)d_in[5];
  const float* c3b   = (const float*)d_in[6];
  const float* w_key = (const float*)d_in[7];
  const float* w_val = (const float*)d_in[8];
  const float* w_ih  = (const float*)d_in[9];    // [768,4352]
  const float* w_hh  = (const float*)d_in[10];   // [768,256]
  const float* b_ih  = (const float*)d_in[11];
  const float* b_hh  = (const float*)d_in[12];
  const float* w_hid = (const float*)d_in[13];   // [256,256]
  const float* adv_w = (const float*)d_in[14];
  const float* adv_b = (const float*)d_in[15];
  const float* val_w = (const float*)d_in[16];
  const float* val_b = (const float*)d_in[17];
  float* out = (float*)d_out;

  // ---- workspace layout (256B aligned) ----
  char* ws = (char*)d_ws;
  size_t cur = 0;
  auto alloc = [&](size_t bytes) { char* p = ws + cur; cur = (cur + bytes + 255) & ~(size_t)255; return p; };
  bf16*  feat  = (bf16*)alloc((size_t)ROWS * FEAT * 2);        // 16 MB
  bf16*  wpack = (bf16*)alloc((size_t)G1N * FEAT * 2);         // 10 MB
  bf16*  wihr  = (bf16*)alloc((size_t)768 * HID * 2);
  bf16*  whhb  = (bf16*)alloc((size_t)768 * HID * 2);
  bf16*  whidb = (bf16*)alloc((size_t)HID * HID * 2);
  float* g1out = (float*)alloc((size_t)ROWS * G1N * 4);        // 10 MB
  bf16*  hseq  = (bf16*)alloc((size_t)ROWS * HID * 2);
  float* oseq  = (float*)alloc((size_t)ROWS * HID * 4);
  float* gbuf  = (float*)alloc((size_t)ROWS * HID * 4);
  bf16*  wpc1  = (bf16*)alloc((size_t)32 * 192 * 2);
  bf16*  wpc2  = (bf16*)alloc((size_t)64 * 1024 * 2);
  bf16*  wpc3  = (bf16*)alloc((size_t)64 * 704 * 2);
  const int CH = 256;                                          // conv chunk (imgs)
  bf16*  c1out = (bf16*)alloc((size_t)CH * 32 * 1521 * 2);     // 25 MB
  bf16*  c2out = (bf16*)alloc((size_t)CH * 64 * 324 * 2);      // 11 MB
  (void)ws_size; (void)in_sizes; (void)n_in; (void)out_size;

  // ---- pack dense-layer weights to bf16 ----
  auto cvt = [&](bf16* dst, const float* src, int rows, int cols, int ld, int off) {
    int tot = rows * cols;
    cvt_bf16<<<(tot + 255) / 256, 256, 0, stream>>>(dst, src, rows, cols, ld, off);
  };
  cvt(wpack,              w_key, 256, FEAT, FEAT, 0);
  cvt(wpack + 256 * FEAT, w_val, 256, FEAT, FEAT, 0);
  cvt(wpack + 512 * FEAT, w_ih,  768, FEAT, FEAT + HID, 0);     // W_ih[:, :4096]
  cvt(wihr,               w_ih,  768, HID,  FEAT + HID, FEAT);  // W_ih[:, 4096:]
  cvt(whhb,               w_hh,  768, HID,  HID, 0);
  cvt(whidb,              w_hid, HID, HID,  HID, 0);

  // ---- pack conv weights (im2col, zero-padded K groups) ----
  pack_conv_w<3, 7, 1, 64><<<(32 * 192 + 255) / 256, 256, 0, stream>>>(wpc1, c1w, 32);
  pack_conv_w<32, 5, 1, 32><<<(64 * 1024 + 255) / 256, 256, 0, stream>>>(wpc2, c2w, 64);
  pack_conv_w<64, 3, 3, 32><<<(64 * 704 + 255) / 256, 256, 0, stream>>>(wpc3, c3w, 64);

  // ---- conv pipeline (implicit-GEMM WMMA), chunked over images ----
  const size_t lds1 = (size_t)3 * 84 * 84 * 2;    // 42 KB
  const size_t lds2 = (size_t)32 * 39 * 39 * 2;   // 97 KB (CDNA5: 320KB/WGP)
  const size_t lds3 = (size_t)64 * 18 * 18 * 2;   // 41 KB
  for (int c = 0; c < ROWS / CH; ++c) {
    const float* in0 = obs + (size_t)c * CH * 3 * 84 * 84;
    conv_wmma<true, 3, 84, 84, 7, 1, 64, 32, 39, 39>
        <<<CH, 256, lds1, stream>>>(in0, wpc1, c1b, c1out);
    conv_wmma<false, 32, 39, 39, 5, 1, 32, 64, 18, 18>
        <<<CH, 256, lds2, stream>>>(c1out, wpc2, c2b, c2out);
    conv_wmma<false, 64, 18, 18, 3, 3, 32, 64, 8, 8>
        <<<CH, 256, lds3, stream>>>(c2out, wpc3, c3b, feat + (size_t)c * CH * FEAT);
  }

  // ---- GEMM1: feat @ [w_key; w_val; W_ih_x]^T -> [2048,1280] ----
  gemm_bf16_nt<<<dim3(ROWS / 128, G1N / 64), 256, 0, stream>>>(
      feat, wpack, g1out, nullptr, ROWS, G1N, FEAT, 0);

  // ---- sequential GRU + attention scan (1 WGP, 32 waves, ~200KB LDS) ----
  const size_t smem = (size_t)(BB * HID * 4)
                    + (size_t)(BB * HID * 2) * 2
                    + (size_t)(BB * 768 * 4)
                    + (size_t)(BB * HID * 4)
                    + (size_t)(BB * SS * 4);
  scan_gru_attn<<<1, 1024, smem, stream>>>(g1out, wihr, whhb, b_ih, b_hh, hseq, oseq);

  // ---- GEMM2: g = relu(hseq @ w_hidden^T + oseq) ----
  gemm_bf16_nt<<<dim3(ROWS / 128, HID / 64), 256, 0, stream>>>(
      hseq, whidb, gbuf, oseq, ROWS, HID, HID, 1);

  // ---- dueling heads ----
  heads_q<<<(ROWS + 255) / 256, 256, 0, stream>>>(gbuf, adv_w, adv_b, val_w, val_b, out);
}
